// ScamDetectionGNN_46316927320191
// MI455X (gfx1250) — matrix operand order repaired and checked by reading, not compiled, not run on hardware
//
#include <hip/hip_runtime.h>
#include <hip/hip_bf16.h>

typedef __attribute__((ext_vector_type(16))) _Float16 v16h;
typedef __attribute__((ext_vector_type(8)))  float    v8f;

#define C_HID 32

// ---------------- degree / normalization ----------------

__global__ void k_init_deg(float* __restrict__ deg, int n) {
    int i = blockIdx.x * blockDim.x + threadIdx.x;
    if (i < n) deg[i] = 1.0f;  // self-loop contributes 1 to every node's degree
}

__global__ void k_edge_deg(const long long* __restrict__ ei, float* __restrict__ deg, int E) {
    int e = blockIdx.x * blockDim.x + threadIdx.x;
    if (e < E) {
        int d = (int)ei[(size_t)E + e];
        atomicAdd(&deg[d], 1.0f);
    }
}

__global__ void k_rsqrt(float* __restrict__ deg, int n) {
    int i = blockIdx.x * blockDim.x + threadIdx.x;
    if (i < n) deg[i] = rsqrtf(deg[i]);  // deg >= 1 always (self-loops)
}

// ---------------- layer 1 GEMM: h1 = x @ W1  (WMMA f16 -> f32) ----------------
// One wave computes a 16x32 output tile via two v_wmma_f32_16x16x32_f16.
// Fused epilogue: h1 stored as f16 (gather table, L2-resident, half traffic);
// acc1 (f32) initialized with the self-loop term h1 * dinv^2.

__global__ __launch_bounds__(256) void k_gemm1(const float* __restrict__ x,
                                               const float* __restrict__ W1,
                                               const float* __restrict__ dinv,
                                               _Float16* __restrict__ h1h,
                                               float* __restrict__ acc1,
                                               int n) {
    const int lane = threadIdx.x & 31;
    const int tile = blockIdx.x * (blockDim.x >> 5) + (threadIdx.x >> 5);
    const int numTiles = (n + 15) >> 4;
    if (tile >= numTiles) return;          // wave-uniform: EXEC stays all-ones for WMMA

    const int row0 = tile << 4;
    const int half = lane >> 4;            // 0: lanes 0-15, 1: lanes 16-31
    const int l15  = lane & 15;

    // ---- A operand: 16x32 f16. Lane l<16 row=l holds K{0..15};
    //      lane l>=16 row=l-16 holds K{16..31}.
    int arow  = row0 + l15;
    int asafe = arow < n ? arow : (n - 1);
    const float* xr = x + (size_t)asafe * C_HID;
    const int kb = half * 16;
    v16h a;
#pragma unroll
    for (int j = 0; j < 8; ++j) {
        a[j]     = (_Float16)xr[kb + j];
        a[j + 8] = (_Float16)xr[8 + kb + j];
    }
    if (arow >= n) {
#pragma unroll
        for (int j = 0; j < 16; ++j) a[j] = (_Float16)0.0f;
    }

    // ---- B operands: 32x16 f16, col N = lane&15; lanes 0-15 hold K0..15,
    //      lanes 16-31 hold K16..31.
    const int koff = half * 16;
    v16h b0, b1;
#pragma unroll
    for (int j = 0; j < 16; ++j) {
        b0[j] = (_Float16)W1[(koff + j) * C_HID + l15];
        b1[j] = (_Float16)W1[(koff + j) * C_HID + 16 + l15];
    }

    v8f c0 = {}, c1 = {};
    c0 = __builtin_amdgcn_wmma_f32_16x16x32_f16(false, a, false, b0, (short)0, c0, false, false);
    c1 = __builtin_amdgcn_wmma_f32_16x16x32_f16(false, a, false, b1, (short)0, c1, false, false);

    // ---- D layout: VGPR v -> row row0 + half*8 + v, col = lane&15
    if (row0 + 16 <= n) {
        // full tile (always taken for n % 16 == 0): wave-uniform, branch-free stores
        const float4* dv = (const float4*)(dinv + row0 + half * 8);
        float4 da = dv[0], db = dv[1];
        float sc[8] = { da.x * da.x, da.y * da.y, da.z * da.z, da.w * da.w,
                        db.x * db.x, db.y * db.y, db.z * db.z, db.w * db.w };
#pragma unroll
        for (int v = 0; v < 8; ++v) {
            int row = row0 + half * 8 + v;
            size_t base = (size_t)row * C_HID + l15;
            h1h[base]        = (_Float16)c0[v];
            h1h[base + 16]   = (_Float16)c1[v];
            acc1[base]       = c0[v] * sc[v];
            acc1[base + 16]  = c1[v] * sc[v];
        }
    } else {
#pragma unroll
        for (int v = 0; v < 8; ++v) {
            int row = row0 + half * 8 + v;
            if (row < n) {
                float di = dinv[row];
                float s  = di * di;
                size_t base = (size_t)row * C_HID + l15;
                h1h[base]       = (_Float16)c0[v];
                h1h[base + 16]  = (_Float16)c1[v];
                acc1[base]      = c0[v] * s;
                acc1[base + 16] = c1[v] * s;
            }
        }
    }
}

// ---------------- layer 1 edge scatter: wave per edge, lane per channel ----------------

__global__ __launch_bounds__(256) void k_scatter1(const long long* __restrict__ ei,
                                                  const float* __restrict__ dinv,
                                                  const _Float16* __restrict__ h1h,
                                                  float* __restrict__ acc1,
                                                  int E) {
    const int lane = threadIdx.x & 31;
    const int e = blockIdx.x * (blockDim.x >> 5) + (threadIdx.x >> 5);
    if (e >= E) return;
    int s = (int)ei[e];
    int d = (int)ei[(size_t)E + e];
    float w = dinv[s] * dinv[d];
    float val = (float)h1h[(size_t)s * C_HID + lane] * w;  // coalesced 64B f16 gather (L2-resident)
    atomicAdd(&acc1[(size_t)d * C_HID + lane], val);       // coalesced 128B atomic burst
}

// ---------------- in-place relu(acc1 + b1) ----------------

__global__ void k_relu_bias(float* __restrict__ acc1, const float* __restrict__ b1, int total) {
    int t = blockIdx.x * blockDim.x + threadIdx.x;
    if (t < total) {
        float v = acc1[t] + b1[t & (C_HID - 1)];
        acc1[t] = v > 0.0f ? v : 0.0f;
    }
}

// ---------------- layer 2 GEMM (32->2) + self-loop, scalar (memory bound) ----------------

__global__ void k_gemm2(const float* __restrict__ h2, const float* __restrict__ W2,
                        const float* __restrict__ dinv, float* __restrict__ gbuf,
                        float* __restrict__ acc2, int n) {
    int i = blockIdx.x * blockDim.x + threadIdx.x;
    if (i >= n) return;
    const float4* hr = (const float4*)(h2 + (size_t)i * C_HID);
    float g0 = 0.0f, g1 = 0.0f;
#pragma unroll
    for (int q = 0; q < 8; ++q) {
        float4 v = hr[q];
        int k = q * 4;
        g0 += v.x * W2[2 * k]     + v.y * W2[2 * (k + 1)]     + v.z * W2[2 * (k + 2)]     + v.w * W2[2 * (k + 3)];
        g1 += v.x * W2[2 * k + 1] + v.y * W2[2 * (k + 1) + 1] + v.z * W2[2 * (k + 2) + 1] + v.w * W2[2 * (k + 3) + 1];
    }
    gbuf[2 * i] = g0;
    gbuf[2 * i + 1] = g1;
    float di = dinv[i];
    float s = di * di;
    acc2[2 * i] = g0 * s;
    acc2[2 * i + 1] = g1 * s;
}

__global__ void k_scatter2(const long long* __restrict__ ei, const float* __restrict__ dinv,
                           const float* __restrict__ gbuf, float* __restrict__ acc2, int E) {
    int e = blockIdx.x * blockDim.x + threadIdx.x;
    if (e >= E) return;
    int s = (int)ei[e];
    int d = (int)ei[(size_t)E + e];
    float w = dinv[s] * dinv[d];
    atomicAdd(&acc2[2 * d],     gbuf[2 * s] * w);
    atomicAdd(&acc2[2 * d + 1], gbuf[2 * s + 1] * w);
}

__global__ void k_out(const float* __restrict__ acc2, const float* __restrict__ b2,
                      float* __restrict__ out, int total) {
    int t = blockIdx.x * blockDim.x + threadIdx.x;
    if (t < total) out[t] = acc2[t] + b2[t & 1];
}

// ---------------- launch ----------------

extern "C" void kernel_launch(void* const* d_in, const int* in_sizes, int n_in,
                              void* d_out, int out_size, void* d_ws, size_t ws_size,
                              hipStream_t stream) {
    const float*     x  = (const float*)d_in[0];
    const long long* ei = (const long long*)d_in[1];  // int64 [2, E]
    const float*     W1 = (const float*)d_in[2];
    const float*     b1 = (const float*)d_in[3];
    const float*     W2 = (const float*)d_in[4];
    const float*     b2 = (const float*)d_in[5];
    float* out = (float*)d_out;

    const int n = in_sizes[0] / C_HID;   // 100000
    const int E = in_sizes[1] / 2;       // 1600000

    // workspace (floats): dinv[n] | acc1[32n] (doubles as h2) | gbuf[2n] | acc2[2n] | h1h[32n f16]
    float* ws   = (float*)d_ws;
    float* dinv = ws;
    float* acc1 = dinv + n;
    float* gbuf = acc1 + (size_t)C_HID * n;
    float* acc2 = gbuf + (size_t)2 * n;
    _Float16* h1h = (_Float16*)(acc2 + (size_t)2 * n);

    const int B = 256;
    const int numTiles = (n + 15) >> 4;

    k_init_deg<<<(n + B - 1) / B, B, 0, stream>>>(dinv, n);
    k_edge_deg<<<(E + B - 1) / B, B, 0, stream>>>(ei, dinv, E);
    k_rsqrt<<<(n + B - 1) / B, B, 0, stream>>>(dinv, n);

    k_gemm1<<<(numTiles + 7) / 8, B, 0, stream>>>(x, W1, dinv, h1h, acc1, n);
    k_scatter1<<<(E + 7) / 8, B, 0, stream>>>(ei, dinv, h1h, acc1, E);
    k_relu_bias<<<((n * C_HID) + B - 1) / B, B, 0, stream>>>(acc1, b1, n * C_HID);

    k_gemm2<<<(n + B - 1) / B, B, 0, stream>>>(acc1, W2, dinv, gbuf, acc2, n);
    k_scatter2<<<(E + B - 1) / B, B, 0, stream>>>(ei, dinv, gbuf, acc2, E);
    k_out<<<((n * 2) + B - 1) / B, B, 0, stream>>>(acc2, b2, out, n * 2);
}
